// Model_39676907882460
// MI455X (gfx1250) — matrix operand order repaired and checked by reading, not compiled, not run on hardware
//
#include <hip/hip_runtime.h>
#include <math.h>

// ---------------- types / WMMA ----------------
typedef __bf16 bf16;
typedef __attribute__((ext_vector_type(16))) __bf16 v16bf;
typedef __attribute__((ext_vector_type(8)))  __bf16 v8bf;
typedef __attribute__((ext_vector_type(8)))  float  v8f;
typedef unsigned int u32x4 __attribute__((ext_vector_type(4)));
typedef int          i32x4 __attribute__((ext_vector_type(4)));
typedef int          i32x8 __attribute__((ext_vector_type(8)));

#define WMMA_BF16(a, b, c) \
  __builtin_amdgcn_wmma_f32_16x16x32_bf16(false, (a), false, (b), (short)0, (c), false, false)

#ifdef __has_builtin
#if __has_builtin(__builtin_amdgcn_tensor_load_to_lds)
#define HAVE_TDM_BUILTIN 1
#endif
#if __has_builtin(__builtin_amdgcn_s_wait_tensorcnt)
#define HAVE_TENSORCNT_BUILTIN 1
#endif
#endif

__device__ inline v8f zero_v8f() {
  v8f z = {0.f, 0.f, 0.f, 0.f, 0.f, 0.f, 0.f, 0.f};
  return z;
}

// ---------------- geometry ----------------
constexpr int Bb = 2, Hh = 8, Ll = 2048, Dd = 1024, DQK = 64;
constexpr int Mrows = Bb * Hh * Ll;          // 32768 fused (B,H,L) rows

// ---------------- deterministic dropout ----------------
constexpr unsigned KEEP_TH  = 3865470566u;   // 0.9 * 2^32
constexpr float    KEEP_INV = 1.0f / 0.9f;

__device__ inline unsigned hash_u32(unsigned x) {
  x ^= 42u * 0x9E3779B9u;                    // fixed "key 42" seed mix
  x ^= x >> 16; x *= 0x7FEB352Du;
  x ^= x >> 15; x *= 0x846CA68Bu;
  x ^= x >> 16;
  return x;
}

// ---------------- Tensor Data Mover: 2D bf16 tile -> LDS ----------------
// D# per CDNA5 ISA ch.8: group0 = {count, lds_addr, global_addr, type=2},
// group1 = {wg_mask=0, data_size=1(2B), dims/tile dims/stride}, groups 2/3 zero.
__device__ inline void tdm_load_tile_bf16_2d(unsigned lds_addr, const void* gptr,
                                             unsigned dim0, unsigned dim1,
                                             unsigned tile0, unsigned tile1,
                                             unsigned stride0) {
  const unsigned long long ga = (unsigned long long)(size_t)gptr;
  u32x4 g0;
  g0[0] = 1u;                                            // count=1, user D#
  g0[1] = lds_addr;                                      // LDS byte address
  g0[2] = (unsigned)(ga & 0xFFFFFFFFull);                // global_addr[31:0]
  g0[3] = (unsigned)((ga >> 32) & 0x1FFFFFFull) | (2u << 30); // [56:32] | type=2
  i32x8 g1;
  g1[0] = (int)(1u << 16);                               // data_size=1 -> 2 bytes
  g1[1] = (int)((dim0 & 0xFFFFu) << 16);                 // tensor_dim0[15:0]
  g1[2] = (int)(((dim0 >> 16) & 0xFFFFu) | ((dim1 & 0xFFFFu) << 16));
  g1[3] = (int)(((dim1 >> 16) & 0xFFFFu) | ((tile0 & 0xFFFFu) << 16));
  g1[4] = (int)(tile1 & 0xFFFFu);                        // tile_dim1, tile_dim2=0
  g1[5] = (int)stride0;                                  // tensor_dim0_stride[31:0]
  g1[6] = 0;
  g1[7] = 0;
#ifdef HAVE_TDM_BUILTIN
  i32x4 z4 = {0, 0, 0, 0};
#if __clang_major__ >= 23
  i32x8 z8 = {0, 0, 0, 0, 0, 0, 0, 0};
  __builtin_amdgcn_tensor_load_to_lds(g0, g1, z4, z4, z8, 0);
#else
  __builtin_amdgcn_tensor_load_to_lds(g0, g1, z4, z4, 0);
#endif
#else
  asm volatile("tensor_load_to_lds %0, %1" :: "s"(g0), "s"(g1) : "memory");
#endif
}

__device__ inline void tdm_wait0() {
#ifdef HAVE_TENSORCNT_BUILTIN
  __builtin_amdgcn_s_wait_tensorcnt(0);
#else
  asm volatile("s_wait_tensorcnt 0x0" ::: "memory");
#endif
}

// ---------------- fragment loaders (ISA 7.12.2 layouts, wave32) ----------------
// A fragment 16x32 bf16: lane l -> M = l&15, half h = l>>4;
// element e<8 : K = h*8 + e ; element e>=8 : K = 16 + h*8 + (e-8)
__device__ inline v16bf frag_a(const bf16* tile, int ld) {
  const int lane = threadIdx.x & 31;
  const int mm = lane & 15, h = lane >> 4;
  const bf16* p = tile + mm * ld + h * 8;
  v8bf lo = *(const v8bf*)(p);
  v8bf hi = *(const v8bf*)(p + 16);
  v16bf a;
#pragma unroll
  for (int e = 0; e < 8; ++e) { a[e] = lo[e]; a[e + 8] = hi[e]; }
  return a;
}

// B fragment 32x16 bf16 from K-contiguous (per-N-row) storage:
// lane l -> N = l&15, half h = l>>4; element e<8 : K = h*8 + e ; e>=8 : K = 16 + h*8 + (e-8)
__device__ inline v16bf frag_b(const bf16* tile, int ld) {
  const int lane = threadIdx.x & 31;
  const int nn = lane & 15, h = lane >> 4;
  const bf16* p = tile + nn * ld + h * 8;
  v8bf lo = *(const v8bf*)(p);
  v8bf hi = *(const v8bf*)(p + 16);
  v16bf b;
#pragma unroll
  for (int e = 0; e < 8; ++e) { b[e] = lo[e]; b[e + 8] = hi[e]; }
  return b;
}

// ======================================================================
// Kernel 1: Y(bf16, Mx64) = X(f32, Mx1024) @ W(f32, 1024x64) + bias
// 128-row tile per block, 8 waves, wave w -> rows [16w,16w+16)
// ======================================================================
__global__ __launch_bounds__(256)
void proj_kernel(const float* __restrict__ X, const float* __restrict__ W,
                 const float* __restrict__ bias, bf16* __restrict__ Y) {
  __shared__ __align__(16) bf16 As[128 * 32];   // 8 KB, row-major [m][k]
  __shared__ __align__(16) bf16 Bt[64 * 32];    // 4 KB, [n][k] (K-contiguous)

  const int t = threadIdx.x;
  const int wave = t >> 5, lane = t & 31;
  const int h = lane >> 4, nn = lane & 15;
  const int mbase = blockIdx.x * 128;

  v8f acc[4] = {zero_v8f(), zero_v8f(), zero_v8f(), zero_v8f()};

  for (int kb = 0; kb < Dd; kb += 32) {
    // stage A tile 128x32 (f32 -> bf16)
    {
      const int row = t >> 1, c0 = (t & 1) * 16;
      const float* g = X + (size_t)(mbase + row) * Dd + kb + c0;
      float4 f0 = *(const float4*)(g + 0);
      float4 f1 = *(const float4*)(g + 4);
      float4 f2 = *(const float4*)(g + 8);
      float4 f3 = *(const float4*)(g + 12);
      if (kb + 32 < Dd) __builtin_prefetch((const char*)g + 32 * 4, 0, 1);
      bf16* s = As + row * 32 + c0;
      s[0]=(bf16)f0.x;  s[1]=(bf16)f0.y;  s[2]=(bf16)f0.z;  s[3]=(bf16)f0.w;
      s[4]=(bf16)f1.x;  s[5]=(bf16)f1.y;  s[6]=(bf16)f1.z;  s[7]=(bf16)f1.w;
      s[8]=(bf16)f2.x;  s[9]=(bf16)f2.y;  s[10]=(bf16)f2.z; s[11]=(bf16)f2.w;
      s[12]=(bf16)f3.x; s[13]=(bf16)f3.y; s[14]=(bf16)f3.z; s[15]=(bf16)f3.w;
    }
    // stage W tile 32x64 transposed -> Bt[n][k] (f32 -> bf16)
    {
      const int k = t & 31, n0 = (t >> 5) * 8;
      const float* g = W + (size_t)(kb + k) * 64 + n0;
      float4 f0 = *(const float4*)(g + 0);
      float4 f1 = *(const float4*)(g + 4);
      Bt[(n0 + 0) * 32 + k] = (bf16)f0.x;
      Bt[(n0 + 1) * 32 + k] = (bf16)f0.y;
      Bt[(n0 + 2) * 32 + k] = (bf16)f0.z;
      Bt[(n0 + 3) * 32 + k] = (bf16)f0.w;
      Bt[(n0 + 4) * 32 + k] = (bf16)f1.x;
      Bt[(n0 + 5) * 32 + k] = (bf16)f1.y;
      Bt[(n0 + 6) * 32 + k] = (bf16)f1.z;
      Bt[(n0 + 7) * 32 + k] = (bf16)f1.w;
    }
    __syncthreads();

    const v16bf a = frag_a(As + (wave * 16) * 32, 32);
#pragma unroll
    for (int n = 0; n < 4; ++n) {
      const v16bf b = frag_b(Bt + (n * 16) * 32, 32);
      acc[n] = WMMA_BF16(a, b, acc[n]);
    }
    __syncthreads();
  }

  // epilogue: + bias, store bf16 (C layout: row = 16*wave + r + 8h, col = 16n + nn)
#pragma unroll
  for (int n = 0; n < 4; ++n) {
    const float bv = bias[n * 16 + nn];
#pragma unroll
    for (int r = 0; r < 8; ++r) {
      const int row = mbase + wave * 16 + r + 8 * h;
      Y[(size_t)row * 64 + n * 16 + nn] = (bf16)(acc[n][r] + bv);
    }
  }
}

// ======================================================================
// Kernel 2: flash attention per (b,h): O = dropout(softmax(Q K^T * s)) V
// grid = (L/128 q-tiles, B*H). Block 256 threads / 8 waves; wave owns 16 rows.
// Q and K tiles are fetched by the Tensor Data Mover; V is staged through
// VGPRs because it needs an LDS transpose.
// ======================================================================
__global__ __launch_bounds__(256)
void attn_kernel(const bf16* __restrict__ Q, const bf16* __restrict__ K,
                 const bf16* __restrict__ V, bf16* __restrict__ Oout,
                 float scale) {
  __shared__ __align__(16) bf16 Qs[128 * 64];   // 16 KB
  __shared__ __align__(16) bf16 Ks[64 * 64];    //  8 KB row-major [key][dim]
  __shared__ __align__(16) bf16 Vt[64 * 64];    //  8 KB transposed [dv][key]
  __shared__ __align__(16) bf16 Ps[8][16 * 64]; // 16 KB per-wave P tiles

  const int t = threadIdx.x;
  const int wave = t >> 5, lane = t & 31;
  const int h = lane >> 4, nn = lane & 15;
  const int bh = blockIdx.y;
  const size_t rowbase = (size_t)bh * Ll + (size_t)blockIdx.x * 128;
  const size_t kvbase  = (size_t)bh * Ll;

  // ---- Q tile 128x64 via TDM ----
  if (wave == 0) {
    tdm_load_tile_bf16_2d((unsigned)(size_t)(void*)Qs, Q + rowbase * 64,
                          /*dim0=*/64, /*dim1=*/(unsigned)Ll,
                          /*tile0=*/64, /*tile1=*/128, /*stride0=*/64);
    tdm_wait0();
  }
  __syncthreads();

  const v16bf qa0 = frag_a(Qs + (wave * 16) * 64 + 0, 64);
  const v16bf qa1 = frag_a(Qs + (wave * 16) * 64 + 32, 64);

  float mrow[8], lrow[8];
  v8f O[4] = {zero_v8f(), zero_v8f(), zero_v8f(), zero_v8f()};
#pragma unroll
  for (int r = 0; r < 8; ++r) { mrow[r] = -3.0e38f; lrow[r] = 0.f; }

  for (int kt = 0; kt < Ll; kt += 64) {
    __syncthreads();   // Ks/Vt reuse across iterations
    // ---- K tile 64x64 via TDM (overlaps with V staging below) ----
    if (wave == 0) {
      tdm_load_tile_bf16_2d((unsigned)(size_t)(void*)Ks,
                            K + (kvbase + kt) * 64,
                            /*dim0=*/64, /*dim1=*/(unsigned)Ll,
                            /*tile0=*/64, /*tile1=*/64, /*stride0=*/64);
    }
    // ---- stage V tile transposed: Vt[dv][key] ----
    {
      const int key = t >> 2, c0 = (t & 3) * 16;
      const bf16* g = V + (kvbase + kt + key) * 64 + c0;
      v8bf a0 = *(const v8bf*)(g);
      v8bf a1 = *(const v8bf*)(g + 8);
#pragma unroll
      for (int e = 0; e < 8; ++e) {
        Vt[(c0 + e) * 64 + key]     = a0[e];
        Vt[(c0 + 8 + e) * 64 + key] = a1[e];
      }
      if (kt + 64 < Ll) __builtin_prefetch((const char*)g + 64 * 64 * 2, 0, 1);
    }
    if (wave == 0) tdm_wait0();
    __syncthreads();

    // ---- S = scale * Q K^T (16 rows x 64 keys per wave) ----
    v8f S[4];
#pragma unroll
    for (int n = 0; n < 4; ++n) {
      v8f z = zero_v8f();
      const v16bf b0 = frag_b(Ks + (n * 16) * 64 + 0, 64);
      const v16bf b1 = frag_b(Ks + (n * 16) * 64 + 32, 64);
      z = WMMA_BF16(qa0, b0, z);
      z = WMMA_BF16(qa1, b1, z);
#pragma unroll
      for (int r = 0; r < 8; ++r) z[r] *= scale;
      S[n] = z;
    }

    // ---- online softmax: row max over 4 n-tiles then across 16-lane half ----
#pragma unroll
    for (int r = 0; r < 8; ++r) {
      float v = fmaxf(fmaxf(S[0][r], S[1][r]), fmaxf(S[2][r], S[3][r]));
      v = fmaxf(v, __shfl_xor(v, 1, 32));
      v = fmaxf(v, __shfl_xor(v, 2, 32));
      v = fmaxf(v, __shfl_xor(v, 4, 32));
      v = fmaxf(v, __shfl_xor(v, 8, 32));
      const float mnew = fmaxf(mrow[r], v);
      const float f = __expf(mrow[r] - mnew);
      mrow[r] = mnew;
      lrow[r] *= f;
#pragma unroll
      for (int n = 0; n < 4; ++n) O[n][r] *= f;
    }

    // ---- P = exp(S - m); dropout on numerator only; stage to LDS ----
    float psum[8];
#pragma unroll
    for (int r = 0; r < 8; ++r) psum[r] = 0.f;
    const unsigned flatbase = (unsigned)bh * (unsigned)(Ll * Ll);
#pragma unroll
    for (int n = 0; n < 4; ++n) {
#pragma unroll
      for (int r = 0; r < 8; ++r) {
        const float p = __expf(S[n][r] - mrow[r]);
        psum[r] += p;
        const unsigned qrow = (unsigned)(blockIdx.x * 128 + wave * 16 + r + 8 * h);
        const unsigned idx  = flatbase + qrow * (unsigned)Ll +
                              (unsigned)(kt + n * 16 + nn);
        const float pd = (hash_u32(idx) <= KEEP_TH) ? p * KEEP_INV : 0.f;
        Ps[wave][(r + 8 * h) * 64 + n * 16 + nn] = (bf16)pd;
      }
    }
#pragma unroll
    for (int r = 0; r < 8; ++r) {
      float v = psum[r];
      v += __shfl_xor(v, 1, 32);
      v += __shfl_xor(v, 2, 32);
      v += __shfl_xor(v, 4, 32);
      v += __shfl_xor(v, 8, 32);
      lrow[r] += v;
    }

    // ---- O += P V (same-wave LDS RAW is in-order) ----
    const v16bf pa0 = frag_a(&Ps[wave][0], 64);
    const v16bf pa1 = frag_a(&Ps[wave][32], 64);
#pragma unroll
    for (int n = 0; n < 4; ++n) {
      const v16bf vb0 = frag_b(Vt + (n * 16) * 64 + 0, 64);
      const v16bf vb1 = frag_b(Vt + (n * 16) * 64 + 32, 64);
      O[n] = WMMA_BF16(pa0, vb0, O[n]);
      O[n] = WMMA_BF16(pa1, vb1, O[n]);
    }
  }

  // epilogue: divide by softmax denominator, store bf16
#pragma unroll
  for (int r = 0; r < 8; ++r) {
    const float inv = 1.0f / lrow[r];
    const size_t row = rowbase + wave * 16 + r + 8 * h;
#pragma unroll
    for (int n = 0; n < 4; ++n)
      Oout[row * 64 + n * 16 + nn] = (bf16)(O[n][r] * inv);
  }
}

// ======================================================================
// Kernel 3: out(f32, Mx1024) = A(bf16, Mx64) @ Wo(f32, 64x1024) + bo
// grid = (M/128, 1024/64)
// ======================================================================
__global__ __launch_bounds__(256)
void oproj_kernel(const bf16* __restrict__ A, const float* __restrict__ Wo,
                  const float* __restrict__ bo, float* __restrict__ Y) {
  __shared__ __align__(16) bf16 As[128 * 64];  // 16 KB
  __shared__ __align__(16) bf16 Bt[64 * 64];   //  8 KB, [n][k]

  const int t = threadIdx.x;
  const int wave = t >> 5, lane = t & 31;
  const int h = lane >> 4, nn = lane & 15;
  const int mbase = blockIdx.x * 128;
  const int nbase = blockIdx.y * 64;

  // stage A tile 128x64
  {
    const int row = t >> 1, c0 = (t & 1) * 32;
    const uint4* g = (const uint4*)(A + (size_t)(mbase + row) * 64 + c0);
    uint4* s = (uint4*)(As + row * 64 + c0);
    s[0] = g[0]; s[1] = g[1]; s[2] = g[2]; s[3] = g[3];
  }
  // stage Wo tile 64x64 transposed (f32 -> bf16)
  {
    const int k = t >> 2, c0 = (t & 3) * 16;
    const float* g = Wo + (size_t)k * Dd + nbase + c0;
    float4 f0 = *(const float4*)(g + 0);
    float4 f1 = *(const float4*)(g + 4);
    float4 f2 = *(const float4*)(g + 8);
    float4 f3 = *(const float4*)(g + 12);
    Bt[(c0 + 0)  * 64 + k] = (bf16)f0.x;  Bt[(c0 + 1)  * 64 + k] = (bf16)f0.y;
    Bt[(c0 + 2)  * 64 + k] = (bf16)f0.z;  Bt[(c0 + 3)  * 64 + k] = (bf16)f0.w;
    Bt[(c0 + 4)  * 64 + k] = (bf16)f1.x;  Bt[(c0 + 5)  * 64 + k] = (bf16)f1.y;
    Bt[(c0 + 6)  * 64 + k] = (bf16)f1.z;  Bt[(c0 + 7)  * 64 + k] = (bf16)f1.w;
    Bt[(c0 + 8)  * 64 + k] = (bf16)f2.x;  Bt[(c0 + 9)  * 64 + k] = (bf16)f2.y;
    Bt[(c0 + 10) * 64 + k] = (bf16)f2.z;  Bt[(c0 + 11) * 64 + k] = (bf16)f2.w;
    Bt[(c0 + 12) * 64 + k] = (bf16)f3.x;  Bt[(c0 + 13) * 64 + k] = (bf16)f3.y;
    Bt[(c0 + 14) * 64 + k] = (bf16)f3.z;  Bt[(c0 + 15) * 64 + k] = (bf16)f3.w;
  }
  __syncthreads();

  const v16bf a0 = frag_a(As + (wave * 16) * 64 + 0, 64);
  const v16bf a1 = frag_a(As + (wave * 16) * 64 + 32, 64);

  v8f acc[4] = {zero_v8f(), zero_v8f(), zero_v8f(), zero_v8f()};
#pragma unroll
  for (int n = 0; n < 4; ++n) {
    const v16bf b0 = frag_b(Bt + (n * 16) * 64 + 0, 64);
    const v16bf b1 = frag_b(Bt + (n * 16) * 64 + 32, 64);
    acc[n] = WMMA_BF16(a0, b0, acc[n]);
    acc[n] = WMMA_BF16(a1, b1, acc[n]);
  }

  // epilogue: + bo, coalesced f32 stores
#pragma unroll
  for (int n = 0; n < 4; ++n) {
    const float bv = bo[nbase + n * 16 + nn];
#pragma unroll
    for (int r = 0; r < 8; ++r) {
      const size_t row = (size_t)(mbase + wave * 16 + r + 8 * h);
      Y[row * Dd + nbase + n * 16 + nn] = acc[n][r] + bv;
    }
  }
}

// ======================================================================
extern "C" void kernel_launch(void* const* d_in, const int* in_sizes, int n_in,
                              void* d_out, int out_size, void* d_ws, size_t ws_size,
                              hipStream_t stream) {
  (void)in_sizes; (void)n_in; (void)out_size; (void)ws_size;
  const float* query = (const float*)d_in[0];
  const float* key   = (const float*)d_in[1];
  const float* value = (const float*)d_in[2];
  const float* Wq = (const float*)d_in[3];
  const float* bq = (const float*)d_in[4];
  const float* Wk = (const float*)d_in[5];
  const float* bk = (const float*)d_in[6];
  const float* Wv = (const float*)d_in[7];
  const float* bv = (const float*)d_in[8];
  const float* Wo = (const float*)d_in[9];
  const float* bo = (const float*)d_in[10];
  float* out = (float*)d_out;

  // workspace layout (bf16): q | k | v | attn_out, each 32768x64 = 4 MB
  const size_t seg = (size_t)Mrows * 64 * sizeof(bf16);
  bf16* qws = (bf16*)d_ws;
  bf16* kws = (bf16*)((char*)d_ws + 1 * seg);
  bf16* vws = (bf16*)((char*)d_ws + 2 * seg);
  bf16* aws = (bf16*)((char*)d_ws + 3 * seg);

  const dim3 blk(256);
  proj_kernel<<<dim3(Mrows / 128), blk, 0, stream>>>(query, Wq, bq, qws);
  proj_kernel<<<dim3(Mrows / 128), blk, 0, stream>>>(key,   Wk, bk, kws);
  proj_kernel<<<dim3(Mrows / 128), blk, 0, stream>>>(value, Wv, bv, vws);

  // dk = (L * dqk / L) / sqrt(d_model) = 64 / 32 = 2.0
  const float scale = ((float)Ll * (float)DQK / (float)Ll) / sqrtf((float)Dd);
  attn_kernel<<<dim3(Ll / 128, Bb * Hh), blk, 0, stream>>>(qws, kws, vws, aws, scale);

  oproj_kernel<<<dim3(Mrows / 128, Dd / 64), blk, 0, stream>>>(aws, Wo, bo, out);
}
